// spatial_attention_38208029065261
// MI455X (gfx1250) — matrix operand order repaired and checked by reading, not compiled, not run on hardware
//
#include <hip/hip_runtime.h>

// ---------------------------------------------------------------------------
// MI455X / gfx1250 fused relative-position spatial attention.
//
// Shapes: BATCH=4800, C_IN=256, T=1, V=25, DK=DV=256, NH=8, DKH=DVH=32.
//
// Roofline: HBM floor = x(123MB)+out(123MB) ~ 10.6us @ 23.3TB/s. The two
// dense GEMMs (qkv proj 47 GF + out proj 16 GF) run on
// v_wmma_f32_16x16x32_bf16 (f32 accumulate). Attention core (3 GF,
// data-dependent rel indexing) runs in f32 VALU out of LDS.
//
// NB=4 batches per workgroup cuts bf16-weight L2 traffic 4x (629MB total vs
// 250MB HBM stream) and lets each A-fragment load pair feed 8 WMMAs.
//
// Round-3 change: round 2 fully unrolled the k-loop, so LICM hoisted all
// 8x8 B-fragments (512 VGPRs) out of the M-tile loop -> scratch spills
// (scratch_load_b128 reloads between WMMAs). The k-loops are now forced to
// `#pragma unroll 1`, keeping live state ~160 VGPRs: acc(64) + A(8) +
// transient B(64). B-fragments are re-read from LDS per M-tile instead
// (cheap), and weights still stream from L2 exactly once per workgroup.
// ---------------------------------------------------------------------------

typedef __bf16 bf16_t;
typedef __attribute__((ext_vector_type(16))) __bf16 v16bf;
typedef __attribute__((ext_vector_type(8)))  __bf16 v8bf;
typedef __attribute__((ext_vector_type(8)))  float  v8f;

#define NBATCH   4800
#define NB       4     // batches per workgroup
#define CIN      256
#define VPOS     25
#define QKV_ROWS 768   // 2*DK + DV
#define NH       8
#define DH       32    // DKH == DVH
#define DOUT     256
#define QSTR     110   // qkv LDS row pitch (220B -> bank step 55, coprime 64)

// ---------------------------------------------------------------------------
// Kernel 0: one-time (per launch) f32 -> bf16 weight conversion into
// workspace. w_qkv: 768x256, w_out: 256x256 -> 512 KiB, L2-resident for all
// 1200 workgroups of the main kernel.
// ---------------------------------------------------------------------------
__global__ __launch_bounds__(256)
void convert_weights_kernel(const float* __restrict__ w_qkv,
                            const float* __restrict__ w_out,
                            bf16_t* __restrict__ wq_bf,
                            bf16_t* __restrict__ wo_bf) {
  int i = blockIdx.x * blockDim.x + threadIdx.x;
  if (i < QKV_ROWS * CIN) wq_bf[i] = (bf16_t)w_qkv[i];
  if (i < DOUT * DOUT)    wo_bf[i] = (bf16_t)w_out[i];
}

// ---------------------------------------------------------------------------
// Fused kernel: grid = 1200 workgroups x 256 threads (8 wave32), 4 batches
// per workgroup, fully resident in LDS (zero intermediate HBM traffic).
// ---------------------------------------------------------------------------
__global__ __launch_bounds__(256)
void fused_spatial_attention_kernel(const float* __restrict__ x,
                                    const float* __restrict__ b_qkv,
                                    const float* __restrict__ b_out,
                                    const float* __restrict__ Amat,
                                    const float* __restrict__ mask,
                                    const float* __restrict__ key_rel,
                                    const float* __restrict__ key_rel_diag,
                                    const bf16_t* __restrict__ wq_bf,
                                    const bf16_t* __restrict__ wo_bf,
                                    float* __restrict__ out) {
  // stageB: WMMA B-matrix for GEMM1 (x^T, bf16) and later GEMM2 (attn, bf16).
  // Rows 25..31 of each batch slice are zero N-padding for both uses.
  __shared__ __align__(64) bf16_t stageB[NB][32][CIN];      // 65,536 B
  __shared__ __align__(64) bf16_t qkvb[QKV_ROWS][QSTR];     // 168,960 B
  __shared__ __align__(64) float  wts[NH][VPOS][26];        //  20,800 B

  const int b0   = blockIdx.x * NB;
  const int tid  = threadIdx.x;
  const int lane = tid & 31;
  const int wave = tid >> 5;

  // Common WMMA fragment-addressing terms (A/B use this code's contiguous-K order, same for both; C is the ISA layout):
  // A 16x32 bf16: lane L holds row M=L&15; K 0..15 for L<16,
  //               16..31 for L>=16  -> two 16B global loads per lane.
  // B 32x16 bf16: lane L holds column N=L&15; 16 consecutive K values
  //               (0..15 for L<16, 16..31 for L>=16) -> one 32B LDS read.
  // C 16x16 f32:  VGPR g -> row M = g (+8 for upper half-wave), col N = L&15.
  const int asel = (lane < 16) ? 0 : 16;  // A K-chunk select
  const int boff = (lane < 16) ? 0 : 16;  // B K-range select
  const int col  = lane & 15;             // N column / A row within tile
  const int rofs = (lane < 16) ? 0 : 8;   // C row offset

  // ---- Phase 0: stage x for 4 batches, float4 loads, transpose+cvt bf16 ----
  {
    const float4* x4 = (const float4*)(x + (size_t)b0 * CIN * VPOS);
    for (int i = tid; i < NB * CIN * VPOS / 4; i += 256) {
      float4 f = x4[i];
      float vals[4] = {f.x, f.y, f.z, f.w};
      const int e0 = i * 4;
#pragma unroll
      for (int j = 0; j < 4; ++j) {
        const int e  = e0 + j;
        const int nb = e / (CIN * VPOS);
        const int r  = e % (CIN * VPOS);
        stageB[nb][r % VPOS][r / VPOS] = (bf16_t)vals[j];
      }
    }
    // zero the N-padding rows (25..31); they stay zero through phase 4 too.
#pragma unroll
    for (int nb = 0; nb < NB; ++nb)
      for (int r = VPOS; r < 32; ++r)
        stageB[nb][r][tid] = (bf16_t)0.0f;
  }
  __syncthreads();

  // ---- Phase 1: QKV GEMM: (768x256) @ (256 x 4*32pad), bf16 WMMA -----------
  // 48 M-tiles x 8 N-tiles x 8 K-steps = 3072 WMMAs/workgroup.
#pragma unroll 1
  for (int mt = wave; mt < QKV_ROWS / 16; mt += 8) {
    const int mbase = mt * 16;
    const bf16_t* arow = wq_bf + (size_t)(mbase + col) * CIN;
    v8f acc[8];
#pragma unroll
    for (int nt = 0; nt < 8; ++nt) acc[nt] = (v8f){};
#pragma unroll 1
    for (int kt = 0; kt < CIN / 32; ++kt) {
      const int kb = kt * 32;
      union { v16bf v; v8bf h[2]; } a;
      a.h[0] = *(const v8bf*)(arow + kb + asel);
      a.h[1] = *(const v8bf*)(arow + kb + 8 + asel);
#pragma unroll
      for (int nt = 0; nt < 8; ++nt) {
        v16bf bv = *(const v16bf*)&stageB[nt >> 1][((nt & 1) << 4) + col][kb + boff];
        acc[nt] = __builtin_amdgcn_wmma_f32_16x16x32_bf16(false, a.v, false, bv,
                                                          (short)0, acc[nt],
                                                          false, false);
      }
    }
    // C -> LDS as bf16 (+ b_qkv bias; zeros in reference, kept for fidelity)
#pragma unroll
    for (int g = 0; g < 8; ++g) {
      const int o = mbase + rofs + g;
      const float bias = b_qkv[o];
#pragma unroll
      for (int nt = 0; nt < 8; ++nt) {
        const int v = ((nt & 1) << 4) + col;
        if (v < VPOS)
          qkvb[o][(nt >> 1) * 27 + v] = (bf16_t)(acc[nt][g] + bias);
      }
    }
  }
  __syncthreads();

  // ---- Phases 2-4 per local batch: logits(+rel) -> softmax -> P@V ----------
  const float scale = 0.17677669529663687f; // 32^-0.5
  for (int nb = 0; nb < NB; ++nb) {
    const int cb = nb * 27; // column base in qkvb

    // logits with folded relative term: rel_to_abs skew collapses to
    // r(m,n) = key_rel[m*24 + (n - (n>m))] (n!=m), key_rel_diag (n==m);
    // logits = s*q.(k+r) + A*mask.
    for (int idx = tid; idx < NH * VPOS * VPOS; idx += 256) {
      const int h = idx / (VPOS * VPOS);
      const int r = idx % (VPOS * VPOS);
      const int m = r / VPOS, n = r % VPOS;
      const int qb = h * DH;
      const int kb = 256 + h * DH;
      const float* rk = (n == m)
          ? key_rel_diag
          : key_rel + (size_t)(m * (VPOS - 1) + (n > m ? n - 1 : n)) * DH;
      float s = 0.0f;
#pragma unroll
      for (int d = 0; d < DH; ++d)
        s += (float)qkvb[qb + d][cb + m] * ((float)qkvb[kb + d][cb + n] + rk[d]);
      wts[h][m][n] = s * scale + Amat[m * VPOS + n] * mask[m * VPOS + n];
    }
    __syncthreads();

    // softmax over n (200 rows, one thread per row)
    if (tid < NH * VPOS) {
      const int h = tid / VPOS, m = tid % VPOS;
      float* row = wts[h][m];
      float mx = row[0];
#pragma unroll
      for (int n = 1; n < VPOS; ++n) mx = fmaxf(mx, row[n]);
      float sum = 0.0f;
#pragma unroll
      for (int n = 0; n < VPOS; ++n) {
        float e = __expf(row[n] - mx);
        row[n] = e;
        sum += e;
      }
      const float inv = 1.0f / sum;
#pragma unroll
      for (int n = 0; n < VPOS; ++n) row[n] *= inv;
    }
    __syncthreads();

    // attn = weights @ v^T, written straight into GEMM2 B-layout:
    // channel = h*32+d, position = m -> stageB[nb][m][h*32+d].
    for (int idx = tid; idx < NH * VPOS * DH; idx += 256) { // 6400 = 25 iters
      const int h = idx / (VPOS * DH);
      const int r = idx % (VPOS * DH);
      const int m = r / DH, d = r % DH;
      const float* row = wts[h][m];
      const int vb = 512 + h * DH + d;
      float s = 0.0f;
#pragma unroll
      for (int n = 0; n < VPOS; ++n) s += row[n] * (float)qkvb[vb][cb + n];
      stageB[nb][m][h * DH + d] = (bf16_t)s;
    }
    __syncthreads();
  }

  // ---- Phase 5: out GEMM: (256x256) @ (256 x 4*32pad), bf16 WMMA -----------
  // 16 M-tiles x 8 N-tiles x 8 K-steps = 1024 WMMAs/workgroup.
#pragma unroll 1
  for (int mt = wave; mt < DOUT / 16; mt += 8) {
    const int mbase = mt * 16;
    const bf16_t* arow = wo_bf + (size_t)(mbase + col) * DOUT;
    v8f acc[8];
#pragma unroll
    for (int nt = 0; nt < 8; ++nt) acc[nt] = (v8f){};
#pragma unroll 1
    for (int kt = 0; kt < DOUT / 32; ++kt) {
      const int kb = kt * 32;
      union { v16bf v; v8bf h[2]; } a;
      a.h[0] = *(const v8bf*)(arow + kb + asel);
      a.h[1] = *(const v8bf*)(arow + kb + 8 + asel);
#pragma unroll
      for (int nt = 0; nt < 8; ++nt) {
        v16bf bv = *(const v16bf*)&stageB[nt >> 1][((nt & 1) << 4) + col][kb + boff];
        acc[nt] = __builtin_amdgcn_wmma_f32_16x16x32_bf16(false, a.v, false, bv,
                                                          (short)0, acc[nt],
                                                          false, false);
      }
    }
#pragma unroll
    for (int g = 0; g < 8; ++g) {
      const int o = mbase + rofs + g;
      const float bias = b_out[o];
#pragma unroll
      for (int nt = 0; nt < 8; ++nt) {
        const int v = ((nt & 1) << 4) + col;
        if (v < VPOS)
          out[((size_t)(b0 + (nt >> 1)) * DOUT + o) * VPOS + v] =
              acc[nt][g] + bias;
      }
    }
  }
}

// ---------------------------------------------------------------------------
// Host-side launcher. Workspace use: 768*256 + 256*256 bf16 = 512 KiB.
// ---------------------------------------------------------------------------
extern "C" void kernel_launch(void* const* d_in, const int* in_sizes, int n_in,
                              void* d_out, int out_size, void* d_ws, size_t ws_size,
                              hipStream_t stream) {
  const float* x            = (const float*)d_in[0];
  const float* w_qkv        = (const float*)d_in[1];
  const float* b_qkv        = (const float*)d_in[2];
  const float* w_out        = (const float*)d_in[3];
  const float* b_out        = (const float*)d_in[4];
  const float* Amat         = (const float*)d_in[5];
  const float* mask         = (const float*)d_in[6];
  const float* key_rel      = (const float*)d_in[7];
  const float* key_rel_diag = (const float*)d_in[8];

  bf16_t* wq_bf = (bf16_t*)d_ws;
  bf16_t* wo_bf = wq_bf + QKV_ROWS * CIN;

  convert_weights_kernel<<<(QKV_ROWS * CIN + 255) / 256, 256, 0, stream>>>(
      w_qkv, w_out, wq_bf, wo_bf);

  fused_spatial_attention_kernel<<<NBATCH / NB, 256, 0, stream>>>(
      x, b_qkv, b_out, Amat, mask, key_rel, key_rel_diag,
      wq_bf, wo_bf, (float*)d_out);
}
